// ScaledDotProductAttention_83820581749084
// MI455X (gfx1250) — matrix operand order repaired and checked by reading, not compiled, not run on hardware
//
#include <hip/hip_runtime.h>

typedef __attribute__((ext_vector_type(16))) _Float16 v16h;
typedef __attribute__((ext_vector_type(8)))  float    v8f;

#define SEQ 2048
#define DK  128
#define NQT (SEQ / 16)                     // 128 query tiles per (b,h)
#define NBH 64                             // 4 batches * 16 heads
#define SOFT_SCALE 0.08838834764831845f    // 1/sqrt(128)
#define VT_STRIDE 48                       // padded key-stride (halves) of V^T in LDS
#define VT_STRIDE32 (VT_STRIDE / 2)        // same, in dwords

__device__ __forceinline__ void ld8(const float* p, float* f) {
  const float4* q = (const float4*)p;
  float4 a = q[0]; float4 b = q[1];
  f[0]=a.x; f[1]=a.y; f[2]=a.z; f[3]=a.w;
  f[4]=b.x; f[5]=b.y; f[6]=b.z; f[7]=b.w;
}

// pack two f32 -> one dword of two f16 (compiler emits v_cvt_pk_f16_f32, RNE)
__device__ __forceinline__ unsigned pkh(float a, float b) {
  union { _Float16 h[2]; unsigned u; } t;
  t.h[0] = (_Float16)a;
  t.h[1] = (_Float16)b;
  return t.u;
}

__device__ __forceinline__ v8f wmma16(v16h a, v16h b, v8f c) {
  // D = A(16x32 f16) * B(32x16 f16) + C(16x16 f32)
  return __builtin_amdgcn_wmma_f32_16x16x32_f16(false, a, false, b,
                                                (short)0, c, false, false);
}

__global__ __launch_bounds__(32)
void fa_fwd_causal(const float* __restrict__ Qg, const float* __restrict__ Kg,
                   const float* __restrict__ Vg, float* __restrict__ Og) {
  // V tile transposed: Vt[d][key], d = 0..127, key = 0..31 (padded to 48 halves)
  __shared__ _Float16 Vt[DK * VT_STRIDE];   // 12 KB

  const int lane = threadIdx.x;
  const int col  = lane & 15;   // q column (N) for S^T; also A-row index for loads
  const int h    = lane >> 4;   // lane half

  const int bh     = blockIdx.x >> 7;          // / NQT
  const int qt     = blockIdx.x & (NQT - 1);
  const int q_base = qt << 4;

  const size_t base = (size_t)bh * SEQ * DK;
  const float* Qp = Qg + base;
  const float* Kp = Kg + base;
  const float* Vp = Vg + base;
  float*       Op = Og + base;

  // ---- Q^T as B-operand fragments (4 chunks of head-dim 32), pre-scaled ----
  // B layout: lane holds column N=col; element i -> K = {0..7|16..23} + 8*h (head-dim)
  v16h qf[4];
  {
    const float* qrow = Qp + (size_t)(q_base + col) * DK;
    #pragma unroll
    for (int c = 0; c < 4; ++c) {
      float f[16];
      ld8(qrow + c*32 + 8*h,      f);
      ld8(qrow + c*32 + 16 + 8*h, f + 8);
      #pragma unroll
      for (int i = 0; i < 16; ++i) qf[c][i] = (_Float16)(f[i] * SOFT_SCALE);
    }
  }

  v8f  O[8] = {};              // 16q x 128d accumulator (8 C-fragments)
  float m = -1e30f, l = 0.0f;  // online-softmax stats for q row (q_base+col)

  const int nb = (q_base + 47) >> 5;   // ceil((q_base+16)/32) key blocks
  for (int jb = 0; jb < nb; ++jb) {
    const int j = jb << 5;

    // ---- prefetch next 32-key K/V tiles (128 cachelines each tile pair) ----
    if (jb + 1 < nb) {
      const char* nk = (const char*)(Kp + (size_t)(j + 32) * DK);
      const char* nv = (const char*)(Vp + (size_t)(j + 32) * DK);
      #pragma unroll
      for (int i = 0; i < 4; ++i) {
        __builtin_prefetch(nk + (size_t)(i*32 + lane) * 128, 0, 3);
        __builtin_prefetch(nv + (size_t)(i*32 + lane) * 128, 0, 3);
      }
    }

    // ---- stage V[j..j+31][0..127] transposed into LDS as packed f16 pairs ----
    __syncthreads();   // WAR vs previous iteration's fragment reads
    {
      const int d0 = lane << 2;
      unsigned* Vt32 = (unsigned*)Vt;
      #pragma unroll 4
      for (int it = 0; it < 32; it += 2) {
        int k0 = j + it;     k0 = k0 < SEQ ? k0 : SEQ - 1;
        int k1 = j + it + 1; k1 = k1 < SEQ ? k1 : SEQ - 1;
        float4 a = *(const float4*)(Vp + (size_t)k0 * DK + d0);
        float4 b = *(const float4*)(Vp + (size_t)k1 * DK + d0);
        const int u = it >> 1;   // dword column holds keys (it, it+1)
        Vt32[(d0 + 0) * VT_STRIDE32 + u] = pkh(a.x, b.x);
        Vt32[(d0 + 1) * VT_STRIDE32 + u] = pkh(a.y, b.y);
        Vt32[(d0 + 2) * VT_STRIDE32 + u] = pkh(a.z, b.z);
        Vt32[(d0 + 3) * VT_STRIDE32 + u] = pkh(a.w, b.w);
      }
    }
    __syncthreads();   // RAW before fragment reads

    // ---- S^T = K * Q^T : two 16-key tiles, 4 head-dim chunks each ----
    v8f st[2] = {};
    #pragma unroll
    for (int kt = 0; kt < 2; ++kt) {
      int key = j + kt*16 + col; key = key < SEQ ? key : SEQ - 1;
      const float* krow = Kp + (size_t)key * DK;
      #pragma unroll
      for (int c = 0; c < 4; ++c) {
        // A layout: lane half h, reg pair v: head-dim = c*32 + {0..7|16..23} + 8h
        float f[16];
        ld8(krow + c*32 + 8*h,      f);
        ld8(krow + c*32 + 16 + 8*h, f + 8);
        v16h ka;
        #pragma unroll
        for (int i = 0; i < 16; ++i) ka[i] = (_Float16)f[i];
        st[kt] = wmma16(ka, qf[c], st[kt]);
      }
    }

    // ---- causal mask: C elem (reg r) is key row j + kt*16 + r + 8h ----
    const int qrow = q_base + col;
    #pragma unroll
    for (int kt = 0; kt < 2; ++kt)
      #pragma unroll
      for (int r = 0; r < 8; ++r) {
        int key = j + kt*16 + r + 8*h;
        st[kt][r] = (key > qrow) ? -1e30f : st[kt][r];
      }

    // ---- online softmax: keys live in regs + lane halves (one shfl_xor) ----
    float mx = -1e30f;
    #pragma unroll
    for (int r = 0; r < 8; ++r) mx = fmaxf(mx, fmaxf(st[0][r], st[1][r]));
    mx = fmaxf(mx, __shfl_xor(mx, 16, 32));
    const float mn  = fmaxf(m, mx);
    const float fac = __expf(m - mn);
    float rs = 0.0f;
    #pragma unroll
    for (int kt = 0; kt < 2; ++kt)
      #pragma unroll
      for (int r = 0; r < 8; ++r) {
        float p = __expf(st[kt][r] - mn);
        st[kt][r] = p;
        rs += p;
      }
    rs += __shfl_xor(rs, 16, 32);
    l = l * fac + rs;
    m = mn;

    // ---- P as A-operand (16q x 32keys): purely lane-local repack ----
    // A pair reg v<4 needs keys 2v+8h,+1 == st0 regs 2v,2v+1 (both halves);
    // v>=4 needs keys 16+2(v-4)+8h,+1 == st1 regs 2(v-4),2(v-4)+1.
    v16h ap;
    #pragma unroll
    for (int i = 0; i < 8; ++i) {
      ap[i]     = (_Float16)st[0][i];
      ap[i + 8] = (_Float16)st[1][i];
    }

    // ---- rescale O (rows of O live in reg index -> broadcast fac by row) ----
    #pragma unroll
    for (int r = 0; r < 8; ++r) {
      float fr = __shfl(fac, r + 8*h, 32);
      #pragma unroll
      for (int fc = 0; fc < 8; ++fc) O[fc][r] *= fr;
    }

    // ---- O += P * V : V B-fragment = 2x ds_load_b128 from transposed LDS ----
    #pragma unroll
    for (int fc = 0; fc < 8; ++fc) {
      union { uint4 u[2]; v16h v; } vb;
      const uint4* p = (const uint4*)(&Vt[(fc*16 + col) * VT_STRIDE + 8*h]);
      vb.u[0] = p[0];
      vb.u[1] = p[2];
      O[fc] = wmma16(ap, vb.v, O[fc]);
    }
  }

  // ---- epilogue: normalize by row-sum and store fp32 ----
  #pragma unroll
  for (int r = 0; r < 8; ++r) {
    const float lr  = __shfl(l, r + 8*h, 32);
    const float inv = 1.0f / lr;
    float* orow = Op + (size_t)(q_base + r + 8*h) * DK;
    #pragma unroll
    for (int fc = 0; fc < 8; ++fc) orow[fc*16 + col] = O[fc][r] * inv;
  }
}

extern "C" void kernel_launch(void* const* d_in, const int* in_sizes, int n_in,
                              void* d_out, int out_size, void* d_ws, size_t ws_size,
                              hipStream_t stream) {
  (void)in_sizes; (void)n_in; (void)out_size; (void)d_ws; (void)ws_size;
  const float* Q = (const float*)d_in[0];
  const float* K = (const float*)d_in[1];
  const float* V = (const float*)d_in[2];
  float*       O = (float*)d_out;
  // one wave (32 threads) per 16-row query tile; 64 (b,h) * 128 tiles = 8192 blocks
  fa_fwd_causal<<<dim3(NBH * NQT), dim3(32), 0, stream>>>(Q, K, V, O);
}
